// XPAttentionLayer_89945205113578
// MI455X (gfx1250) — matrix operand order repaired and checked by reading, not compiled, not run on hardware
//
#include <hip/hip_runtime.h>
#include <hip/hip_bf16.h>

// ---------------------------------------------------------------------------
// XPAttentionLayer for MI455X (gfx1250).
//  - All GEMMs: V_WMMA_F32_16X16X32_BF16 (f32 accumulation).
//  - All large intermediates stored bf16 in HBM (halves the dominant traffic;
//    workload is HBM-bound: ~42 GFLOP vs ~1 GB traffic at 23.3 TB/s).
//  - Channel-mix GEMM double-buffers its B-slab in LDS with
//    global_load_async_to_lds_b128 + partial s_wait_asynccnt (ASYNCcnt
//    pipelining); falls back to plain LDS copy if the builtin is absent.
// ---------------------------------------------------------------------------

typedef __attribute__((ext_vector_type(16))) __bf16 v16bf;
typedef __attribute__((ext_vector_type(8)))  __bf16 v8bf;
typedef __attribute__((ext_vector_type(8)))  float  v8f;
typedef __attribute__((ext_vector_type(4)))  int    v4i;

#define B_    2
#define C_    96
#define H_    256
#define W_    256
#define HW_   (H_ * W_)
#define D3_   288
#define HEAD_ 8
#define DH_   32           // d = W / head
#define BH_   16           // b * head
#define EPS_  1e-12f

#if defined(__has_builtin)
#  if __has_builtin(__builtin_amdgcn_global_load_async_to_lds_b128)
#    define HAVE_ASYNC 1
#  endif
#endif
#ifndef HAVE_ASYNC
#  define HAVE_ASYNC 0
#endif

#if defined(__has_builtin) && __has_builtin(__builtin_amdgcn_s_wait_asynccnt)
#  define ASYNC_WAIT(n) __builtin_amdgcn_s_wait_asynccnt(n)
#else
#  define ASYNC_WAIT(n) asm volatile("s_wait_asynccnt %0" ::"n"(n) : "memory")
#endif

// ---- WMMA wrapper ----------------------------------------------------------
__device__ __forceinline__ v8f wmma_bf16(v16bf a, v16bf b, v8f c) {
  return __builtin_amdgcn_wmma_f32_16x16x32_bf16(false, a, false, b,
                                                 (short)0, c, false, false);
}

// ---- Fragment loaders (wave32; contiguous k = 16*(lane/16) + i in A and B) -------------------
// A: 16(M) x 32(K) from f32 row-major (weights; tiny, stays f32)
__device__ __forceinline__ v16bf load_a_f32(const float* __restrict__ A, int lda) {
  int lane = threadIdx.x & 31;
  int m  = lane & 15;
  int kb = (lane >> 4) << 4;
  v16bf r;
#pragma unroll
  for (int i = 0; i < 16; ++i) {
    int k = kb + i;
    r[i] = (__bf16)A[m * lda + k];
  }
  return r;
}

// A: 16(M) x 32(K) from bf16 row-major: two contiguous 16B chunks per lane
__device__ __forceinline__ v16bf load_a_bf(const __bf16* __restrict__ A, int lda) {
  int lane = threadIdx.x & 31;
  int m  = lane & 15;
  int kb = (lane >> 4) << 4;
  v8bf lo = *(const v8bf*)(A + (size_t)m * lda + kb);
  v8bf hi = *(const v8bf*)(A + (size_t)m * lda + kb + 8);
  v16bf r;
#pragma unroll
  for (int i = 0; i < 8; ++i) { r[i] = lo[i]; r[i + 8] = hi[i]; }
  return r;
}

// B: 32(K) x 16(N) from f32 row-major (stride ldb between K rows)
__device__ __forceinline__ v16bf load_b_f32(const float* __restrict__ Bp, long long ldb) {
  int lane = threadIdx.x & 31;
  int n  = lane & 15;
  int kb = (lane >> 4) << 4;
  v16bf r;
#pragma unroll
  for (int i = 0; i < 16; ++i)
    r[i] = (__bf16)Bp[(long long)(kb + i) * ldb + n];
  return r;
}

// B: 32(K) x 16(N) from bf16 row-major (strided b16 gathers)
__device__ __forceinline__ v16bf load_b_bf(const __bf16* __restrict__ Bp, long long ldb) {
  int lane = threadIdx.x & 31;
  int n  = lane & 15;
  int kb = (lane >> 4) << 4;
  v16bf r;
#pragma unroll
  for (int i = 0; i < 16; ++i)
    r[i] = Bp[(long long)(kb + i) * ldb + n];
  return r;
}

// B from bf16 memory holding B^T (mem[n*ldt + k]): per-lane contiguous 32B
__device__ __forceinline__ v16bf load_bt_bf(const __bf16* __restrict__ Bp, int ldt) {
  int lane = threadIdx.x & 31;
  int n  = lane & 15;
  int kb = (lane >> 4) << 4;
  v8bf lo = *(const v8bf*)(Bp + (size_t)n * ldt + kb);
  v8bf hi = *(const v8bf*)(Bp + (size_t)n * ldt + kb + 8);
  v16bf r;
#pragma unroll
  for (int i = 0; i < 8; ++i) { r[i] = lo[i]; r[i + 8] = hi[i]; }
  return r;
}

// C/D stores
__device__ __forceinline__ void store_c_f32(float* __restrict__ Cp, long long ldc, v8f acc) {
  int lane = threadIdx.x & 31;
  int n  = lane & 15;
  int mb = (lane >> 4) << 3;
#pragma unroll
  for (int r = 0; r < 8; ++r)
    Cp[(long long)(mb + r) * ldc + n] = acc[r];
}

__device__ __forceinline__ void store_c_bf(__bf16* __restrict__ Cp, long long ldc, v8f acc) {
  int lane = threadIdx.x & 31;
  int n  = lane & 15;
  int mb = (lane >> 4) << 3;
#pragma unroll
  for (int r = 0; r < 8; ++r)
    Cp[(long long)(mb + r) * ldc + n] = (__bf16)acc[r];
}

// ---------------------------------------------------------------------------
// Stage 1: qkv = qkv_w (288x96) @ x (96 x HW) per batch -> bf16
// ---------------------------------------------------------------------------
__global__ __launch_bounds__(256) void k_qkv(const float* __restrict__ x,
                                             const float* __restrict__ w,
                                             __bf16* __restrict__ qkv) {
  int wv = threadIdx.x >> 5;
  int n0 = (blockIdx.x * 8 + wv) * 16;
  int m0 = blockIdx.y * 16;
  int b  = blockIdx.z;
  const float* xb = x + (long long)b * C_ * HW_;
  v8f acc = {};
#pragma unroll
  for (int kk = 0; kk < C_; kk += 32) {
    v16bf a  = load_a_f32(w + m0 * C_ + kk, C_);
    v16bf bb = load_b_f32(xb + (long long)kk * HW_ + n0, HW_);
    acc = wmma_bf16(a, bb, acc);
  }
  store_c_bf(qkv + (long long)b * D3_ * HW_ + (long long)m0 * HW_ + n0, HW_, acc);
}

// ---------------------------------------------------------------------------
// Stage 2: depthwise 3x3 (SAME) + head split + L2 norm (q,k) over d
// ---------------------------------------------------------------------------
__global__ __launch_bounds__(256) void k_dw_split(const __bf16* __restrict__ qkv,
                                                  const float* __restrict__ dw,
                                                  __bf16* __restrict__ qn,
                                                  __bf16* __restrict__ kn,
                                                  __bf16* __restrict__ vn) {
  __shared__ float s[HEAD_];
  int h = blockIdx.x, o = blockIdx.y, b = blockIdx.z;
  int wcol = threadIdx.x;
  const __bf16* src = qkv + ((long long)(b * D3_ + o) * H_) * W_;
  float acc = 0.f;
#pragma unroll
  for (int dy = 0; dy < 3; ++dy) {
    int hy = h + dy - 1;
    if (hy < 0 || hy >= H_) continue;
#pragma unroll
    for (int dx = 0; dx < 3; ++dx) {
      int wx = wcol + dx - 1;
      if (wx < 0 || wx >= W_) continue;
      acc += (float)src[hy * W_ + wx] * dw[o * 9 + dy * 3 + dx];
    }
  }
  int g = o / C_, c = o % C_;
  int head = wcol & 7, d = wcol >> 3;
  int bh = b * HEAD_ + head;
  long long di = (((long long)bh * C_ + c) * H_ + h) * DH_ + d;
  if (g == 2) { vn[di] = (__bf16)acc; return; }   // uniform per block
  if (threadIdx.x < HEAD_) s[threadIdx.x] = 0.f;
  __syncthreads();
  atomicAdd(&s[head], acc * acc);                 // ds_add_f32 per-head
  __syncthreads();
  float nrm = sqrtf(s[head]);
  float val = acc / fmaxf(nrm, EPS_);
  (g == 0 ? qn : kn)[di] = (__bf16)val;
}

// ---------------------------------------------------------------------------
// Stage 3: attn = (q . k^T) * temperature per (bh,c): 256x32 @ 32x256 -> bf16
// ---------------------------------------------------------------------------
__global__ __launch_bounds__(256) void k_attn(const __bf16* __restrict__ qn,
                                              const __bf16* __restrict__ kn,
                                              const float* __restrict__ temp,
                                              __bf16* __restrict__ attn) {
  int wv = threadIdx.x >> 5;
  int tile = blockIdx.x * 8 + wv;
  int m0 = (tile >> 4) * 16, n0 = (tile & 15) * 16;
  long long mat = blockIdx.y;                        // bh*C + c
  const __bf16* q = qn + mat * H_ * DH_;
  const __bf16* k = kn + mat * H_ * DH_;
  v16bf a  = load_a_bf(q + m0 * DH_, DH_);
  v16bf bb = load_bt_bf(k + n0 * DH_, DH_);
  v8f acc = {};
  acc = wmma_bf16(a, bb, acc);
  float t = temp[0];
  __bf16* Cp = attn + mat * H_ * H_ + (long long)m0 * H_ + n0;
  int lane = threadIdx.x & 31;
  int n = lane & 15, mb = (lane >> 4) << 3;
#pragma unroll
  for (int r = 0; r < 8; ++r) Cp[(mb + r) * H_ + n] = (__bf16)(acc[r] * t);
}

// ---------------------------------------------------------------------------
// Stage 4: softmax along j (row length 256), f32 math on bf16 storage
// ---------------------------------------------------------------------------
__global__ __launch_bounds__(256) void k_softmax(__bf16* __restrict__ attn) {
  __shared__ float sm[256];
  long long row = (long long)blockIdx.y * H_ + blockIdx.x;
  __bf16* p = attn + row * H_;
  int t = threadIdx.x;
  float v = (float)p[t];
  sm[t] = v; __syncthreads();
#pragma unroll
  for (int s = 128; s > 0; s >>= 1) {
    if (t < s) sm[t] = fmaxf(sm[t], sm[t + s]);
    __syncthreads();
  }
  float mx = sm[0]; __syncthreads();
  float e = __expf(v - mx);
  sm[t] = e; __syncthreads();
#pragma unroll
  for (int s = 128; s > 0; s >>= 1) {
    if (t < s) sm[t] += sm[t + s];
    __syncthreads();
  }
  p[t] = (__bf16)(e / sm[0]);
}

// ---------------------------------------------------------------------------
// Stage 5: channel mix: attn2 = conv_w (96x96) @ attn + bias, per bh.
// B-slab (32K x 128N bf16 = 8KB) double-buffered in LDS via async loads:
//   prologue stage 0; at step i issue stage i+1, then s_wait_asynccnt 2
//   (only stage i must have landed; async loads complete in order).
// ---------------------------------------------------------------------------
__device__ __forceinline__ void stage_tile(const __bf16* __restrict__ Bp,
                                           long long nblk, int kk,
                                           __bf16 (*buf)[136], int tid) {
#pragma unroll
  for (int cch = 0; cch < 2; ++cch) {
    int chunk = tid * 2 + cch;          // 512 x 16B chunks / 256 threads
    int row = chunk >> 4;
    int col = (chunk & 15) * 8;
    const __bf16* gp = Bp + (long long)(kk + row) * HW_ + nblk + col;
    __bf16* lp = &buf[row][col];
#if HAVE_ASYNC
    __builtin_amdgcn_global_load_async_to_lds_b128(
        (__attribute__((address_space(1))) v4i*)gp,
        (__attribute__((address_space(3))) v4i*)lp, 0, 0);
#else
    *(v8bf*)lp = *(const v8bf*)gp;
#endif
  }
}

__global__ __launch_bounds__(256) void k_mix(const __bf16* __restrict__ attn,
                                             const float* __restrict__ cw,
                                             const float* __restrict__ cb,
                                             __bf16* __restrict__ attn2) {
  __shared__ __bf16 tile[2][32][136];              // +8 pad: spread LDS banks
  int tid  = threadIdx.x;
  int wv   = tid >> 5;
  int lane = tid & 31;
  long long nblk = (long long)blockIdx.x * 128;    // block base N
  int m0 = blockIdx.y * 16;
  int bh = blockIdx.z;
  const __bf16* Bp = attn + (long long)bh * C_ * HW_;
  int n  = lane & 15;
  int kb = (lane >> 4) << 4;

  stage_tile(Bp, nblk, 0, tile[0], tid);           // prologue
  v8f acc = {};
  for (int i = 0; i < 3; ++i) {                    // K = 3 x 32
    int kk = i * 32;
    if (i < 2) stage_tile(Bp, nblk, kk + 32, tile[(i + 1) & 1], tid);
#if HAVE_ASYNC
    if (i < 2) ASYNC_WAIT(2); else ASYNC_WAIT(0);  // wait only stage i
#endif
    __syncthreads();                               // all waves' chunks landed
    v16bf a = load_a_f32(cw + m0 * C_ + kk, C_);
    v16bf bb;
#pragma unroll
    for (int ii = 0; ii < 16; ++ii) bb[ii] = tile[i & 1][kb + ii][wv * 16 + n];
    acc = wmma_bf16(a, bb, acc);
    __syncthreads();                               // reads done before reuse
  }
  __bf16* Cp = attn2 + (long long)bh * C_ * HW_ + (long long)m0 * HW_ +
               nblk + wv * 16;
  int mb = (lane >> 4) << 3;
#pragma unroll
  for (int r = 0; r < 8; ++r)
    Cp[(long long)(mb + r) * HW_ + n] = (__bf16)(acc[r] + cb[m0 + mb + r]);
}

// ---------------------------------------------------------------------------
// Stage 6: out = attn2 @ v per (bh,c): 256x256 @ 256x32, fused head-merge
// ---------------------------------------------------------------------------
__global__ __launch_bounds__(256) void k_out(const __bf16* __restrict__ attn2,
                                             const __bf16* __restrict__ vn,
                                             __bf16* __restrict__ outm) {
  int wv = threadIdx.x >> 5;
  int tile = blockIdx.x * 8 + wv;                  // 32 tiles (16 M x 2 N)
  int m0 = (tile >> 1) * 16, n0 = (tile & 1) * 16;
  long long mat = blockIdx.y;                      // bh*C + c
  int bh = (int)(mat / C_), c = (int)(mat % C_);
  int b = bh / HEAD_, head = bh % HEAD_;
  const __bf16* Ap = attn2 + mat * H_ * H_;
  const __bf16* Vp = vn + mat * H_ * DH_;
  v8f acc = {};
#pragma unroll
  for (int kk = 0; kk < H_; kk += 32) {
    if (kk + 32 < H_)
      __builtin_prefetch(Ap + (size_t)m0 * H_ + kk + 32, 0, 0);
    v16bf a  = load_a_bf(Ap + (size_t)m0 * H_ + kk, H_);
    v16bf bb = load_b_bf(Vp + (long long)kk * DH_ + n0, DH_);
    acc = wmma_bf16(a, bb, acc);
  }
  int lane = threadIdx.x & 31;
  int n = lane & 15, mb = (lane >> 4) << 3;
  __bf16* Cp = outm + ((long long)(b * C_ + c) * H_) * W_;
#pragma unroll
  for (int r = 0; r < 8; ++r)
    Cp[(long long)(m0 + mb + r) * W_ + ((n0 + n) * HEAD_ + head)] = (__bf16)acc[r];
}

// ---------------------------------------------------------------------------
// Stage 7: project_out: out(f32) = proj_w (96x96) @ outm, per batch
// ---------------------------------------------------------------------------
__global__ __launch_bounds__(256) void k_proj(const __bf16* __restrict__ outm,
                                              const float* __restrict__ pw,
                                              float* __restrict__ out) {
  int wv = threadIdx.x >> 5;
  long long n0 = ((long long)blockIdx.x * 8 + wv) * 16;
  int m0 = blockIdx.y * 16;
  int b  = blockIdx.z;
  const __bf16* Bp = outm + (long long)b * C_ * HW_;
  v8f acc = {};
#pragma unroll
  for (int kk = 0; kk < C_; kk += 32) {
    v16bf a  = load_a_f32(pw + m0 * C_ + kk, C_);
    v16bf bb = load_b_bf(Bp + (long long)kk * HW_ + n0, HW_);
    acc = wmma_bf16(a, bb, acc);
  }
  store_c_f32(out + (long long)b * C_ * HW_ + (long long)m0 * HW_ + n0, HW_, acc);
}

// ---------------------------------------------------------------------------
extern "C" void kernel_launch(void* const* d_in, const int* in_sizes, int n_in,
                              void* d_out, int out_size, void* d_ws, size_t ws_size,
                              hipStream_t stream) {
  (void)in_sizes; (void)n_in; (void)out_size; (void)ws_size;
  const float* x    = (const float*)d_in[0];
  const float* qkvw = (const float*)d_in[1];
  const float* dww  = (const float*)d_in[2];
  const float* cw   = (const float*)d_in[3];
  const float* cb   = (const float*)d_in[4];
  const float* pw   = (const float*)d_in[5];
  const float* temp = (const float*)d_in[6];
  float* out = (float*)d_out;

  __bf16* ws = (__bf16*)d_ws;
  const long long N_QKV = (long long)B_ * D3_ * HW_;        // 37,748,736
  const long long N_HD  = (long long)BH_ * C_ * H_ * DH_;   // 12,582,912
  const long long N_ATT = (long long)BH_ * C_ * H_ * H_;    // 100,663,296
  __bf16* qkv   = ws;
  __bf16* qn    = qkv   + N_QKV;
  __bf16* kn    = qn    + N_HD;
  __bf16* vn    = kn    + N_HD;
  __bf16* attn  = vn    + N_HD;
  __bf16* attn2 = attn  + N_ATT;
  __bf16* outm  = attn2 + N_ATT;

  k_qkv     <<<dim3(HW_ / 128, D3_ / 16, B_),        256, 0, stream>>>(x, qkvw, qkv);
  k_dw_split<<<dim3(H_, D3_, B_),                    256, 0, stream>>>(qkv, dww, qn, kn, vn);
  k_attn    <<<dim3(32, BH_ * C_),                   256, 0, stream>>>(qn, kn, temp, attn);
  k_softmax <<<dim3(H_, BH_ * C_),                   256, 0, stream>>>(attn);
  k_mix     <<<dim3(HW_ / 128, C_ / 16, BH_),        256, 0, stream>>>(attn, cw, cb, attn2);
  k_out     <<<dim3(4, BH_ * C_),                    256, 0, stream>>>(attn2, vn, outm);
  k_proj    <<<dim3(HW_ / 128, C_ / 16, B_),         256, 0, stream>>>(outm, pw, out);
}
